// ZINC20PretrainModel_89550068122081
// MI455X (gfx1250) — matrix operand order repaired and checked by reading, not compiled, not run on hardware
//
#include <hip/hip_runtime.h>
#include <hip/hip_bf16.h>
#include <math.h>

typedef __attribute__((ext_vector_type(16))) __bf16 v16bf;
typedef __attribute__((ext_vector_type(8)))  float  v8f;

#define TPB 256

// =====================================================================
// Weight pack: B[K x Nc] f32 row-major -> bf16 fragments.
// Fragment (kt,nt) holds 32 lanes x 16 bf16, element j of lane:
//   col = nt*16 + (lane&15),  k = kt*32 + (j&7) + (j>=8 ? 16 : 0) + (lane>=16 ? 8 : 0)
// stored contiguously: Bp[((kt*ntN + nt)*32 + lane)*16 + j]
// =====================================================================
__global__ void k_pack_b(const float* __restrict__ B, int K, int Nc,
                         __bf16* __restrict__ Bp)
{
  int ntN = (Nc + 15) >> 4;
  int ntK = (K + 31) >> 5;
  long long total = (long long)ntK * ntN * 512;
  long long t = (long long)blockIdx.x * blockDim.x + threadIdx.x;
  if (t >= total) return;
  int j    = (int)(t & 15);
  int lane = (int)((t >> 4) & 31);
  int ft   = (int)(t >> 9);
  int nt = ft % ntN, kt = ft / ntN;
  int col = (nt << 4) + (lane & 15);
  int k   = (kt << 5) + (j & 7) + ((j >> 3) << 4) + ((lane >> 4) << 3);
  float v = (k < K && col < Nc) ? B[(size_t)k * Nc + col] : 0.0f;
  Bp[t] = (__bf16)v;
}

// =====================================================================
// WMMA bf16 GEMM: C[M x Ncols] = A[M x K] * Bp (+bias)(relu)
// One wave32 -> 16 x (16*Q) strip, Q in {4,2,1} chosen so ntN % Q == 0:
// straight-line inner loop, no divergent guards around WMMA.
// M must be a multiple of 16 (100000, 4000 ok). K zero-padded.
// =====================================================================
template <int Q>
__global__ __launch_bounds__(TPB) void k_gemm_bf16(
    const float* __restrict__ A, int lda,
    const __bf16* __restrict__ Bp,
    const float* __restrict__ bias,
    float* __restrict__ C, int ldc,
    int M, int K, int Ncols, int relu)
{
  int ntN  = (Ncols + 15) >> 4;                 // multiple of Q (launcher guarantee)
  int ntNQ = ntN / Q;
  // scalar-pinned wave id -> all tile math & branches are SALU, EXEC stays all-1
  int wave_in_blk = __builtin_amdgcn_readfirstlane((int)threadIdx.x >> 5);
  int wave = blockIdx.x * (TPB >> 5) + wave_in_blk;
  int tiles = (M >> 4) * ntNQ;
  if (wave >= tiles) return;
  int lane = threadIdx.x & 31;
  int tm = wave / ntNQ, ts = wave - tm * ntNQ;
  int nt0 = ts * Q;
  int half = lane >> 4;                         // 0: lanes 0-15, 1: lanes 16-31
  int l16  = lane & 15;
  int ntK  = (K + 31) >> 5;

  v8f acc[Q] = {};
  const float* Arow = A + (size_t)((tm << 4) + l16) * lda;

  for (int kt = 0; kt < ntK; ++kt) {
    int kb = kt << 5;
    v16bf af;
    if (kb + 32 <= K) {
      // fast path: two contiguous 8-float runs per lane -> 4x b128 loads
      const float4* p0 = (const float4*)(Arow + kb + (half << 3));
      const float4* p1 = (const float4*)(Arow + kb + 16 + (half << 3));
      float4 a0 = p0[0], a1 = p0[1], a2 = p1[0], a3 = p1[1];
      af[0]=(__bf16)a0.x;  af[1]=(__bf16)a0.y;  af[2]=(__bf16)a0.z;  af[3]=(__bf16)a0.w;
      af[4]=(__bf16)a1.x;  af[5]=(__bf16)a1.y;  af[6]=(__bf16)a1.z;  af[7]=(__bf16)a1.w;
      af[8]=(__bf16)a2.x;  af[9]=(__bf16)a2.y;  af[10]=(__bf16)a2.z; af[11]=(__bf16)a2.w;
      af[12]=(__bf16)a3.x; af[13]=(__bf16)a3.y; af[14]=(__bf16)a3.z; af[15]=(__bf16)a3.w;
    } else {
      // tail path (layer-1 K=29): clamped, branch-free loads
#pragma unroll
      for (int j = 0; j < 16; ++j) {
        int ka = kb + ((j < 8) ? j : (j + 8)) + (half << 3);
        int kc = (ka < K) ? ka : 0;
        float av = Arow[kc];
        af[j] = (__bf16)((ka < K) ? av : 0.0f);
      }
    }
    const __bf16* bbase = Bp + (((size_t)kt * ntN + nt0) << 9) + (lane << 4);
#pragma unroll
    for (int q = 0; q < Q; ++q) {
      v16bf bfm = *(const v16bf*)(bbase + ((size_t)q << 9));
      acc[q] = __builtin_amdgcn_wmma_f32_16x16x32_bf16(false, af, false, bfm,
                                                       (short)0, acc[q], false, false);
    }
  }

#pragma unroll
  for (int q = 0; q < Q; ++q) {
    int col = ((nt0 + q) << 4) + l16;
    if (col < Ncols) {
      float bb = bias ? bias[col] : 0.0f;
#pragma unroll
      for (int v = 0; v < 8; ++v) {
        int row = (tm << 4) + v + (half << 3);
        float r = acc[q][v] + bb;
        if (relu) r = fmaxf(r, 0.0f);
        C[(size_t)row * ldc + col] = r;
      }
    }
  }
}

// ---------------------------------------------------------------------
__global__ void k_fill(float* __restrict__ p, size_t n, float v) {
  size_t t = (size_t)blockIdx.x * blockDim.x + threadIdx.x;
  if (t < n) p[t] = v;
}

// per-node attention scores: s1[n,h] = <h[n,h,:], a_src[h,:]>, s2 likewise
__global__ void k_node_scores(const float* __restrict__ hlin,
                              const float* __restrict__ asrc,
                              const float* __restrict__ adst,
                              float* __restrict__ ssrc, float* __restrict__ sdst,
                              int N, int H, int C)
{
  int t = blockIdx.x * blockDim.x + threadIdx.x;
  if (t >= N * H) return;
  int n = t / H, h = t - n * H;
  const float* hv = hlin + ((size_t)n * H + h) * C;
  const float* a1 = asrc + h * C;
  const float* a2 = adst + h * C;
  float s1 = 0.f, s2 = 0.f;
  for (int c = 0; c < C; c += 4) {
    float4 x4 = *(const float4*)(hv + c);
    float4 u4 = *(const float4*)(a1 + c);
    float4 w4 = *(const float4*)(a2 + c);
    s1 += x4.x*u4.x + x4.y*u4.y + x4.z*u4.z + x4.w*u4.w;
    s2 += x4.x*w4.x + x4.y*w4.y + x4.z*w4.z + x4.w*w4.w;
  }
  ssrc[t] = s1; sdst[t] = s2;
}

__device__ __forceinline__ void edge_pair(const int* __restrict__ ei, int E, int e,
                                          int& s, int& d)
{
  if (e < E) { s = ei[e]; d = ei[E + e]; }   // edge_index[0]=src, [1]=dst
  else       { s = e - E; d = s; }           // appended self-loops
}

__device__ __forceinline__ float leaky02(float x) { return (x > 0.f) ? x : 0.2f * x; }

// float atomic-max via monotone int mapping
__device__ __forceinline__ void atomicMaxF(float* addr, float val) {
  if (val >= 0.f) atomicMax((int*)addr, __float_as_int(val));
  else            atomicMin((unsigned int*)addr, __float_as_uint(val));
}

__global__ void k_edge_max(const int* __restrict__ ei, int E, int ET, int H,
                           const float* __restrict__ ssrc, const float* __restrict__ sdst,
                           float* __restrict__ mmax)
{
  int t = blockIdx.x * blockDim.x + threadIdx.x;
  if (t >= ET * H) return;
  int e = t / H, h = t - e * H;
  int s, d; edge_pair(ei, E, e, s, d);
  float x = leaky02(ssrc[s * H + h] + sdst[d * H + h]);
  atomicMaxF(&mmax[d * H + h], x);
}

__global__ void k_edge_sum(const int* __restrict__ ei, int E, int ET, int H,
                           const float* __restrict__ ssrc, const float* __restrict__ sdst,
                           const float* __restrict__ mmax, float* __restrict__ denm)
{
  int t = blockIdx.x * blockDim.x + threadIdx.x;
  if (t >= ET * H) return;
  int e = t / H, h = t - e * H;
  int s, d; edge_pair(ei, E, e, s, d);
  float x = leaky02(ssrc[s * H + h] + sdst[d * H + h]);
  atomicAdd(&denm[d * H + h], expf(x - mmax[d * H + h]));
}

__global__ void k_edge_agg(const int* __restrict__ ei, int E, int ET, int H, int C,
                           const float* __restrict__ ssrc, const float* __restrict__ sdst,
                           const float* __restrict__ mmax, const float* __restrict__ denm,
                           const float* __restrict__ hlin, float* __restrict__ outp)
{
  int t = blockIdx.x * blockDim.x + threadIdx.x;
  if (t >= ET * H) return;
  int e = t / H, h = t - e * H;
  int s, d; edge_pair(ei, E, e, s, d);
  float x = leaky02(ssrc[s * H + h] + sdst[d * H + h]);
  float alpha = expf(x - mmax[d * H + h]) / (denm[d * H + h] + 1e-16f);
  const float* hs = hlin + ((size_t)s * H + h) * C;
  float*       od = outp + ((size_t)d * H + h) * C;
  for (int c = 0; c < C; c += 4) {
    float4 v4 = *(const float4*)(hs + c);
    atomicAdd(&od[c + 0], alpha * v4.x);
    atomicAdd(&od[c + 1], alpha * v4.y);
    atomicAdd(&od[c + 2], alpha * v4.z);
    atomicAdd(&od[c + 3], alpha * v4.w);
  }
}

// x = elu( (x + bias - rm) * g/sqrt(rv+eps) + be )   (in place)
__global__ void k_bias_bn_elu(float* __restrict__ x, const float* __restrict__ bias,
                              const float* __restrict__ g, const float* __restrict__ be,
                              const float* __restrict__ rm, const float* __restrict__ rv,
                              int F, size_t total)
{
  size_t t = (size_t)blockIdx.x * blockDim.x + threadIdx.x;
  if (t >= total) return;
  int c = (int)(t % F);
  float v = x[t] + bias[c];
  v = (v - rm[c]) * (g[c] * rsqrtf(rv[c] + 1e-5f)) + be[c];
  x[t] = (v > 0.f) ? v : expm1f(v);
}

// -------- global mean pool --------
__global__ void k_pool_feat(const float* __restrict__ h, const int* __restrict__ batch,
                            float* __restrict__ gemb, int N)
{
  int t = blockIdx.x * blockDim.x + threadIdx.x;
  if (t >= N * 128) return;
  int n = t >> 7, c = t & 127;
  atomicAdd(&gemb[batch[n] * 128 + c], h[t]);
}
__global__ void k_pool_cnt(const int* __restrict__ batch, float* __restrict__ gcnt, int N)
{
  int n = blockIdx.x * blockDim.x + threadIdx.x;
  if (n < N) atomicAdd(&gcnt[batch[n]], 1.0f);
}
__global__ void k_pool_div(float* __restrict__ gemb, const float* __restrict__ gcnt, int G)
{
  int t = blockIdx.x * blockDim.x + threadIdx.x;
  if (t >= G * 128) return;
  gemb[t] /= fmaxf(gcnt[t >> 7], 1.0f);
}

// -------- loss reductions (block reduce + one atomic per block) --------
__device__ __forceinline__ void block_reduce_add(float v, float* acc) {
  __shared__ float sm[TPB];
  sm[threadIdx.x] = v;
  __syncthreads();
  for (int s = TPB / 2; s > 0; s >>= 1) {
    if ((int)threadIdx.x < s) sm[threadIdx.x] += sm[threadIdx.x + s];
    __syncthreads();
  }
  if (threadIdx.x == 0) atomicAdd(acc, sm[0]);
}

__global__ __launch_bounds__(TPB) void k_ctx_loss(const float* __restrict__ z,
                                                  const float* __restrict__ ctx,
                                                  float* __restrict__ acc, int total)
{
  int t = blockIdx.x * blockDim.x + threadIdx.x;
  float v = 0.f;
  if (t < total) {
    float zz = z[t];
    v = fmaxf(zz, 0.f) + log1pf(expf(-fabsf(zz))) - zz * ctx[t];
  }
  block_reduce_add(v, &acc[0]);
}

__global__ __launch_bounds__(TPB) void k_prop_loss(const float* __restrict__ pz,
                                                   const float* __restrict__ props,
                                                   float* __restrict__ acc, int total)
{
  int t = blockIdx.x * blockDim.x + threadIdx.x;
  float v = 0.f;
  if (t < total) { float d = pz[t] - props[t]; v = d * d; }
  block_reduce_add(v, &acc[1]);
}

__global__ __launch_bounds__(TPB) void k_mask_loss(const float* __restrict__ mz,
                                                   const float* __restrict__ x,
                                                   const unsigned char* __restrict__ mask,
                                                   float* __restrict__ acc, int N)
{
  int t = blockIdx.x * blockDim.x + threadIdx.x;
  float v = 0.f;
  if (t < N * 29) {
    int n = t / 29;
    if (mask[n]) { float d = mz[t] - x[t]; v = d * d; }
  }
  block_reduce_add(v, &acc[2]);
}

__global__ __launch_bounds__(TPB) void k_mask_count(const unsigned char* __restrict__ mask,
                                                    float* __restrict__ acc, int N)
{
  int n = blockIdx.x * blockDim.x + threadIdx.x;
  float v = (n < N && mask[n]) ? 1.f : 0.f;
  block_reduce_add(v, &acc[3]);
}

__global__ void k_final(const float* __restrict__ acc, float* __restrict__ out,
                        int N, int G)
{
  if (blockIdx.x == 0 && threadIdx.x == 0) {
    float ctx = acc[0] / (9.0f * (float)N);
    float prp = acc[1] / (9.0f * (float)G);
    float msk = acc[2] / (fmaxf(acc[3], 1.0f) * 29.0f);
    out[0] = ctx + prp + 0.5f * msk;
  }
}

// =====================================================================
extern "C" void kernel_launch(void* const* d_in, const int* in_sizes, int n_in,
                              void* d_out, int out_size, void* d_ws, size_t ws_size,
                              hipStream_t stream)
{
  const int IN = 29, HID = 128;
  const int N = in_sizes[0] / IN;
  const int E = in_sizes[1] / 2;
  const int G = in_sizes[4] / 9;
  const int ET = E + N;              // with self-loops
  const int H4 = 4 * HID;            // 512

  const float* x       = (const float*)d_in[0];
  const int*   ei      = (const int*)d_in[1];
  const int*   batch   = (const int*)d_in[2];
  const float* context = (const float*)d_in[3];
  const float* props   = (const float*)d_in[4];
  const unsigned char* mask = (const unsigned char*)d_in[5];
  const float* W1 = (const float*)d_in[6],  *as1 = (const float*)d_in[7],
             *ad1 = (const float*)d_in[8],  *b1  = (const float*)d_in[9],
             *g1  = (const float*)d_in[10], *be1 = (const float*)d_in[11],
             *rm1 = (const float*)d_in[12], *rv1 = (const float*)d_in[13];
  const float* W2 = (const float*)d_in[14], *as2 = (const float*)d_in[15],
             *ad2 = (const float*)d_in[16], *b2  = (const float*)d_in[17],
             *g2  = (const float*)d_in[18], *be2 = (const float*)d_in[19],
             *rm2 = (const float*)d_in[20], *rv2 = (const float*)d_in[21];
  const float* W3 = (const float*)d_in[22], *as3 = (const float*)d_in[23],
             *ad3 = (const float*)d_in[24], *b3  = (const float*)d_in[25],
             *g3  = (const float*)d_in[26], *be3 = (const float*)d_in[27],
             *rm3 = (const float*)d_in[28], *rv3 = (const float*)d_in[29];
  const float* cw1 = (const float*)d_in[30], *cb1 = (const float*)d_in[31],
             *cw2 = (const float*)d_in[32], *cb2 = (const float*)d_in[33];
  const float* pw1 = (const float*)d_in[34], *pb1 = (const float*)d_in[35],
             *pw2 = (const float*)d_in[36], *pb2 = (const float*)d_in[37];
  const float* mw1 = (const float*)d_in[38], *mb1 = (const float*)d_in[39],
             *mw2 = (const float*)d_in[40], *mb2 = (const float*)d_in[41];
  float* out = (float*)d_out;

  // ---- workspace carve-up ----
  float* ws = (float*)d_ws;
  size_t o = 0;
  float* bufA = ws + o; o += (size_t)N * H4;     // ping
  float* bufB = ws + o; o += (size_t)N * H4;     // pong
  float* bufC = ws + o; o += (size_t)N * H4;     // pong2
  float* ssrc = ws + o; o += (size_t)N * 4;
  float* sdst = ws + o; o += (size_t)N * 4;
  float* mmax = ws + o; o += (size_t)N * 4;
  float* denm = ws + o; o += (size_t)N * 4;
  float* gemb = ws + o; o += (size_t)G * 128;
  float* gcnt = ws + o; o += (size_t)G;
  float* gz1  = ws + o; o += (size_t)G * 128;
  float* gz2  = ws + o; o += (size_t)G * 16;
  float* acc  = ws + o; o += 16;
  // bf16 packed-weight region
  auto packsize = [](int K, int Nc) {
    return (size_t)((K + 31) / 32) * ((Nc + 15) / 16) * 512;
  };
  __bf16* wb = (__bf16*)(ws + o);
  size_t ob = 0;
  __bf16* W1p  = wb + ob; ob += packsize(IN,  H4);
  __bf16* W2p  = wb + ob; ob += packsize(H4,  H4);
  __bf16* W3p  = wb + ob; ob += packsize(H4,  HID);
  __bf16* cw1p = wb + ob; ob += packsize(HID, HID);
  __bf16* cw2p = wb + ob; ob += packsize(HID, 9);
  __bf16* pw1p = wb + ob; ob += packsize(HID, HID);
  __bf16* pw2p = wb + ob; ob += packsize(HID, 9);
  __bf16* mw1p = wb + ob; ob += packsize(HID, HID);
  __bf16* mw2p = wb + ob; ob += packsize(HID, IN);
  (void)ws_size; (void)n_in; (void)out_size;

  auto cdiv = [](long long a, long long b) { return (unsigned)((a + b - 1) / b); };

  auto fill = [&](float* p, size_t n, float v) {
    k_fill<<<cdiv((long long)n, TPB), TPB, 0, stream>>>(p, n, v);
  };
  auto pack = [&](const float* B, int K, int Nc, __bf16* Bp) {
    long long total = (long long)packsize(K, Nc);
    k_pack_b<<<cdiv(total, TPB), TPB, 0, stream>>>(B, K, Nc, Bp);
  };
  auto gemm = [&](const float* A, int lda, const __bf16* Bp,
                  const float* bias, float* C, int ldc,
                  int M, int K, int Ncols, int relu) {
    int ntN = (Ncols + 15) >> 4;
    if ((ntN & 3) == 0) {
      long long tiles = (long long)(M >> 4) * (ntN >> 2);
      k_gemm_bf16<4><<<cdiv(tiles, TPB / 32), TPB, 0, stream>>>(
          A, lda, Bp, bias, C, ldc, M, K, Ncols, relu);
    } else if ((ntN & 1) == 0) {
      long long tiles = (long long)(M >> 4) * (ntN >> 1);
      k_gemm_bf16<2><<<cdiv(tiles, TPB / 32), TPB, 0, stream>>>(
          A, lda, Bp, bias, C, ldc, M, K, Ncols, relu);
    } else {
      long long tiles = (long long)(M >> 4) * ntN;
      k_gemm_bf16<1><<<cdiv(tiles, TPB / 32), TPB, 0, stream>>>(
          A, lda, Bp, bias, C, ldc, M, K, Ncols, relu);
    }
  };

  // ---- pack all weight matrices to bf16 fragment layout (L2-resident) ----
  pack(W1,  IN,  H4,  W1p);  pack(W2,  H4,  H4,  W2p);  pack(W3, H4, HID, W3p);
  pack(cw1, HID, HID, cw1p); pack(cw2, HID, 9,   cw2p);
  pack(pw1, HID, HID, pw1p); pack(pw2, HID, 9,   pw2p);
  pack(mw1, HID, HID, mw1p); pack(mw2, HID, IN,  mw2p);

  auto gat = [&](const float* hin, int Fin, int H, int C,
                 const __bf16* Wp, const float* as_, const float* ad_, const float* bias,
                 const float* g, const float* be, const float* rm, const float* rv,
                 float* hlin, float* outp) {
    int F = H * C;
    gemm(hin, Fin, Wp, nullptr, hlin, F, N, Fin, F, 0);     // h = h_in @ W (WMMA)
    int NH = N * H;
    k_node_scores<<<cdiv(NH, TPB), TPB, 0, stream>>>(hlin, as_, ad_, ssrc, sdst, N, H, C);
    fill(mmax, (size_t)NH, -INFINITY);
    fill(denm, (size_t)NH, 0.f);
    fill(outp, (size_t)N * F, 0.f);
    long long tot = (long long)ET * H;
    k_edge_max<<<cdiv(tot, TPB), TPB, 0, stream>>>(ei, E, ET, H, ssrc, sdst, mmax);
    k_edge_sum<<<cdiv(tot, TPB), TPB, 0, stream>>>(ei, E, ET, H, ssrc, sdst, mmax, denm);
    k_edge_agg<<<cdiv(tot, TPB), TPB, 0, stream>>>(ei, E, ET, H, C, ssrc, sdst, mmax,
                                                   denm, hlin, outp);
    size_t NF = (size_t)N * F;
    k_bias_bn_elu<<<cdiv((long long)NF, TPB), TPB, 0, stream>>>(outp, bias, g, be, rm, rv,
                                                                F, NF);
  };

  // ---- 3 GAT layers (ping-pong through bufA/B/C) ----
  gat(x,    IN, 4, HID, W1p, as1, ad1, b1, g1, be1, rm1, rv1, bufA, bufB); // h1 = bufB
  gat(bufB, H4, 4, HID, W2p, as2, ad2, b2, g2, be2, rm2, rv2, bufA, bufC); // h2 = bufC
  gat(bufC, H4, 1, HID, W3p, as3, ad3, b3, g3, be3, rm3, rv3, bufA, bufB); // h3 = bufB

  const float* h3 = bufB;   // [N,128]

  // ---- global mean pool ----
  fill(gemb, (size_t)G * 128, 0.f);
  fill(gcnt, (size_t)G, 0.f);
  k_pool_feat<<<cdiv((long long)N * 128, TPB), TPB, 0, stream>>>(h3, batch, gemb, N);
  k_pool_cnt<<<cdiv(N, TPB), TPB, 0, stream>>>(batch, gcnt, N);
  k_pool_div<<<cdiv((long long)G * 128, TPB), TPB, 0, stream>>>(gemb, gcnt, G);

  // ---- loss accumulators ----
  fill(acc, 16, 0.f);

  // context head: z = relu(h3@cw1+cb1)@cw2+cb2  -> BCE-with-logits mean
  gemm(h3, HID, cw1p, cb1, bufA, HID, N, HID, HID, 1);
  gemm(bufA, HID, cw2p, cb2, bufC, 9, N, HID, 9, 0);
  k_ctx_loss<<<cdiv((long long)N * 9, TPB), TPB, 0, stream>>>(bufC, context, acc, N * 9);

  // property head on pooled embeddings -> MSE mean
  gemm(gemb, HID, pw1p, pb1, gz1, HID, G, HID, HID, 1);
  gemm(gz1, HID, pw2p, pb2, gz2, 9, G, HID, 9, 0);
  k_prop_loss<<<cdiv((long long)G * 9, TPB), TPB, 0, stream>>>(gz2, props, acc, G * 9);

  // masked reconstruction head -> masked MSE
  gemm(h3, HID, mw1p, mb1, bufA, HID, N, HID, HID, 1);
  gemm(bufA, HID, mw2p, mb2, bufC, IN, N, HID, IN, 0);
  k_mask_loss<<<cdiv((long long)N * IN, TPB), TPB, 0, stream>>>(bufC, x, mask, acc, N);
  k_mask_count<<<cdiv(N, TPB), TPB, 0, stream>>>(mask, acc, N);

  k_final<<<1, 32, 0, stream>>>(acc, out, N, G);
}
